// Block_79018808312215
// MI455X (gfx1250) — compile-verified, matrix operand order checked by reading, not run on hardware
//
#include <hip/hip_runtime.h>
#include <hip/hip_bf16.h>
#include <math.h>

typedef __bf16 bf16_t;
typedef __attribute__((ext_vector_type(16))) __bf16 v16bf;
typedef __attribute__((ext_vector_type(8)))  float  v8f;

#define DDIM 1024
#define TTOK 2048
#define SEQ  1024
#define NH   16
#define NKV  4
#define HDIM 64
#define NE   8
#define FDIM 3584
#define LDW  40   // padded LDS row stride (bf16 elems) -> conflict-free frag reads

// ---------------------------------------------------------------------------
// WMMA fragment helpers (wave32, 16x16x32 bf16, f32 accum)
// A-operand: lane l holds row m=l&15; k-elems: e<8 -> k0+h*8+e, e>=8 -> k0+16+h*8+(e-8)
// B-operand: identical pattern with "row" = output column (so feeding row-major
// W[N,K] as B computes A*W^T).
// ---------------------------------------------------------------------------
__device__ __forceinline__ v16bf load_frag_bf16(const bf16_t* base,
                                                int ld, int row0, int k0, int lane) {
  int r = row0 + (lane & 15);
  int h = lane >> 4;
  const bf16_t* p = base + (size_t)r * ld + k0 + h * 8;
  v16bf f;
#pragma unroll
  for (int e = 0; e < 8; ++e) f[e] = p[e];
#pragma unroll
  for (int e = 0; e < 8; ++e) f[8 + e] = p[16 + e];
  return f;
}

__device__ __forceinline__ v8f wmma_bf16(v16bf a, v16bf b, v8f c) {
  return __builtin_amdgcn_wmma_f32_16x16x32_bf16(false, a, false, b, (short)0, c,
                                                 false, false);
}

__device__ __forceinline__ v8f v8f_zero() {
  v8f z = {0.f, 0.f, 0.f, 0.f, 0.f, 0.f, 0.f, 0.f};
  return z;
}

// Cooperative 64x32 f32 weight tile staging: 256 threads, 8 f32 each.
__device__ __forceinline__ void stage_fetch(const float* __restrict__ W, int K,
                                            int n0, int k0, int tid, float r[8]) {
  int row = tid >> 2;
  int col = (tid & 3) * 8;
  const float* p = W + (size_t)(n0 + row) * K + k0 + col;
#pragma unroll
  for (int e = 0; e < 8; ++e) r[e] = p[e];
}

__device__ __forceinline__ void stage_store(bf16_t* lds, int tid, const float r[8]) {
  int row = tid >> 2;
  int col = (tid & 3) * 8;
  bf16_t* q = lds + row * LDW + col;
#pragma unroll
  for (int e = 0; e < 8; ++e) q[e] = (bf16_t)r[e];
}

// ---------------------------------------------------------------------------
// RMSNorm: one block per token, f32 in -> bf16 out
// ---------------------------------------------------------------------------
__global__ void rmsnorm_kernel(const float* __restrict__ x, const float* __restrict__ w,
                               bf16_t* __restrict__ out) {
  __shared__ float red[8];
  int t = blockIdx.x;
  const float* xr = x + (size_t)t * DDIM;
  float s = 0.f;
  for (int d = threadIdx.x; d < DDIM; d += 256) { float v = xr[d]; s += v * v; }
#pragma unroll
  for (int off = 16; off > 0; off >>= 1) s += __shfl_xor(s, off, 32);
  int wave = threadIdx.x >> 5, lane = threadIdx.x & 31;
  if (lane == 0) red[wave] = s;
  __syncthreads();
  float tot = red[0] + red[1] + red[2] + red[3] + red[4] + red[5] + red[6] + red[7];
  float rs = rsqrtf(tot / (float)DDIM + 1e-5f);
  for (int d = threadIdx.x; d < DDIM; d += 256)
    out[(size_t)t * DDIM + d] = (bf16_t)(xr[d] * rs * w[d]);
}

// ---------------------------------------------------------------------------
// WMMA GEMM with LDS-staged weights: C[M,N] f32 = A[M,K] bf16 * W[N,K]^T
// block 256 thr = 8 waves, wave tile 16x64, block tile 128x64.
// Weight tile (64x32) converted f32->bf16 once per block, double-buffered.
// B-frags batched before the WMMA burst so LDS latency overlaps the matrix pipe.
// ---------------------------------------------------------------------------
__global__ void gemm_kernel(const bf16_t* __restrict__ A, const float* __restrict__ W,
                            float* __restrict__ C, int K, int N) {
  __shared__ bf16_t wt[2][64 * LDW];
  int tid = threadIdx.x;
  int lane = tid & 31, wave = tid >> 5;
  int m0 = blockIdx.y * 128 + wave * 16;
  int n0 = blockIdx.x * 64;
  v8f acc[4] = {v8f_zero(), v8f_zero(), v8f_zero(), v8f_zero()};
  float rg[8];
  stage_fetch(W, K, n0, 0, tid, rg);
  int nk = K >> 5;
  for (int kt = 0; kt < nk; ++kt) {
    int buf = kt & 1;
    stage_store(&wt[buf][0], tid, rg);
    __syncthreads();
    if (kt + 1 < nk) stage_fetch(W, K, n0, (kt + 1) * 32, tid, rg);
    v16bf a = load_frag_bf16(A, K, m0, kt * 32, lane);
    v16bf b0 = load_frag_bf16(&wt[buf][0], LDW, 0, 0, lane);
    v16bf b1 = load_frag_bf16(&wt[buf][0], LDW, 16, 0, lane);
    v16bf b2 = load_frag_bf16(&wt[buf][0], LDW, 32, 0, lane);
    v16bf b3 = load_frag_bf16(&wt[buf][0], LDW, 48, 0, lane);
    acc[0] = wmma_bf16(a, b0, acc[0]);
    acc[1] = wmma_bf16(a, b1, acc[1]);
    acc[2] = wmma_bf16(a, b2, acc[2]);
    acc[3] = wmma_bf16(a, b3, acc[3]);
  }
  int nc = lane & 15, hl = lane >> 4;
#pragma unroll
  for (int nn = 0; nn < 4; ++nn)
#pragma unroll
    for (int r = 0; r < 8; ++r) {
      int m = m0 + r + 8 * hl;
      C[(size_t)m * N + n0 + nn * 16 + nc] = acc[nn][r];
    }
}

// Same + residual add of X (O-projection: hres = attn*wo^T + x)
__global__ void gemm_residual_kernel(const bf16_t* __restrict__ A,
                                     const float* __restrict__ W,
                                     const float* __restrict__ X,
                                     float* __restrict__ C, int K, int N) {
  __shared__ bf16_t wt[2][64 * LDW];
  int tid = threadIdx.x;
  int lane = tid & 31, wave = tid >> 5;
  int m0 = blockIdx.y * 128 + wave * 16;
  int n0 = blockIdx.x * 64;
  v8f acc[4] = {v8f_zero(), v8f_zero(), v8f_zero(), v8f_zero()};
  float rg[8];
  stage_fetch(W, K, n0, 0, tid, rg);
  int nk = K >> 5;
  for (int kt = 0; kt < nk; ++kt) {
    int buf = kt & 1;
    stage_store(&wt[buf][0], tid, rg);
    __syncthreads();
    if (kt + 1 < nk) stage_fetch(W, K, n0, (kt + 1) * 32, tid, rg);
    v16bf a = load_frag_bf16(A, K, m0, kt * 32, lane);
    v16bf b0 = load_frag_bf16(&wt[buf][0], LDW, 0, 0, lane);
    v16bf b1 = load_frag_bf16(&wt[buf][0], LDW, 16, 0, lane);
    v16bf b2 = load_frag_bf16(&wt[buf][0], LDW, 32, 0, lane);
    v16bf b3 = load_frag_bf16(&wt[buf][0], LDW, 48, 0, lane);
    acc[0] = wmma_bf16(a, b0, acc[0]);
    acc[1] = wmma_bf16(a, b1, acc[1]);
    acc[2] = wmma_bf16(a, b2, acc[2]);
    acc[3] = wmma_bf16(a, b3, acc[3]);
  }
  int nc = lane & 15, hl = lane >> 4;
#pragma unroll
  for (int nn = 0; nn < 4; ++nn)
#pragma unroll
    for (int r = 0; r < 8; ++r) {
      int m = m0 + r + 8 * hl;
      size_t idx = (size_t)m * N + n0 + nn * 16 + nc;
      C[idx] = acc[nn][r] + X[idx];
    }
}

// ---------------------------------------------------------------------------
// RoPE + f32->bf16 convert. Layout [T, nh*64]; thread handles dim pair (d,d+32)
// ---------------------------------------------------------------------------
__global__ void rope_kernel(const float* __restrict__ qf, const float* __restrict__ cosb,
                            const float* __restrict__ sinb, bf16_t* __restrict__ qb,
                            int nh) {
  int i = blockIdx.x * 256 + threadIdx.x;
  int total = TTOK * nh * 32;
  if (i >= total) return;
  int d = i & 31;
  int rest = i >> 5;
  int hh = rest % nh;
  int t = rest / nh;
  int s = t & (SEQ - 1);
  size_t base = (size_t)t * nh * HDIM + hh * HDIM;
  float q0 = qf[base + d], q1 = qf[base + d + 32];
  float c0 = cosb[s * HDIM + d], s0 = sinb[s * HDIM + d];
  float c1 = cosb[s * HDIM + d + 32], s1 = sinb[s * HDIM + d + 32];
  qb[base + d]      = (bf16_t)(q0 * c0 - q1 * s0);
  qb[base + d + 32] = (bf16_t)(q1 * c1 + q0 * s1);
}

__global__ void cvt_kernel(const float* __restrict__ a, bf16_t* __restrict__ o, int n) {
  int i = blockIdx.x * 256 + threadIdx.x;
  if (i < n) o[i] = (bf16_t)a[i];
}

__global__ void zero_kernel(float* __restrict__ p, int n) {
  int i = blockIdx.x * 256 + threadIdx.x;
  if (i < n) p[i] = 0.f;
}

__global__ void addout_kernel(const float* __restrict__ a, const float* __restrict__ b,
                              float* __restrict__ o, int n) {
  int i = blockIdx.x * 256 + threadIdx.x;
  if (i < n) o[i] = a[i] + b[i];
}

// ---------------------------------------------------------------------------
// Flash attention (causal, GQA 16q/4kv heads). One wave = 16 q rows.
// grid (S/128, H, B), block 256. WMMA for QK^T and PV; P relaid via wave-
// private LDS with explicit s_wait_dscnt (no barrier -> divergent trip counts OK).
// ---------------------------------------------------------------------------
__global__ void attn_kernel(const bf16_t* __restrict__ qb, const bf16_t* __restrict__ kb,
                            const bf16_t* __restrict__ vb, bf16_t* __restrict__ ob) {
  __shared__ bf16_t pbuf[8][16 * 32];
  int lane = threadIdx.x & 31, wave = threadIdx.x >> 5;
  int h = blockIdx.y, b = blockIdx.z;
  int qabs = blockIdx.x * 128 + wave * 16;
  int kvh = h >> 2;
  const bf16_t* Q  = qb + (size_t)b * SEQ * (NH * HDIM) + h * HDIM;
  const bf16_t* Kx = kb + (size_t)b * SEQ * (NKV * HDIM) + kvh * HDIM;
  const bf16_t* Vx = vb + (size_t)b * SEQ * (NKV * HDIM) + kvh * HDIM;

  v16bf aq0 = load_frag_bf16(Q, NH * HDIM, qabs, 0, lane);
  v16bf aq1 = load_frag_bf16(Q, NH * HDIM, qabs, 32, lane);

  v8f o[4] = {v8f_zero(), v8f_zero(), v8f_zero(), v8f_zero()};
  float mrow[8], lrow[8];
#pragma unroll
  for (int r = 0; r < 8; ++r) { mrow[r] = -1e30f; lrow[r] = 0.f; }
  int nc = lane & 15, hl = lane >> 4;

  int nkt = (qabs + 47) >> 5;  // causal: cover cols up to qabs+15
  for (int kt = 0; kt < nkt; ++kt) {
    int kc = kt * 32;
    v8f sc[2];
#pragma unroll
    for (int sub = 0; sub < 2; ++sub) {
      v16bf bk0 = load_frag_bf16(Kx, NKV * HDIM, kc + sub * 16, 0, lane);
      sc[sub] = wmma_bf16(aq0, bk0, v8f_zero());
      v16bf bk1 = load_frag_bf16(Kx, NKV * HDIM, kc + sub * 16, 32, lane);
      sc[sub] = wmma_bf16(aq1, bk1, sc[sub]);
    }
    float p0s[8], p1s[8];
#pragma unroll
    for (int r = 0; r < 8; ++r) {
      int qg = qabs + r + 8 * hl;
      float v0 = sc[0][r] * 0.125f;
      if (kc + nc > qg) v0 = -1e30f;
      float v1 = sc[1][r] * 0.125f;
      if (kc + 16 + nc > qg) v1 = -1e30f;
      float mx = fmaxf(v0, v1);
#pragma unroll
      for (int off = 1; off < 16; off <<= 1) mx = fmaxf(mx, __shfl_xor(mx, off, 32));
      float mnew = fmaxf(mrow[r], mx);
      float alpha = __expf(mrow[r] - mnew);
      float p0 = __expf(v0 - mnew), p1 = __expf(v1 - mnew);
      float ps = p0 + p1;
#pragma unroll
      for (int off = 1; off < 16; off <<= 1) ps += __shfl_xor(ps, off, 32);
      lrow[r] = lrow[r] * alpha + ps;
      mrow[r] = mnew;
      p0s[r] = p0; p1s[r] = p1;
#pragma unroll
      for (int nn = 0; nn < 4; ++nn) o[nn][r] *= alpha;
    }
    // relayout P (C-layout -> A-layout) through wave-private LDS
#pragma unroll
    for (int r = 0; r < 8; ++r) {
      pbuf[wave][(r + 8 * hl) * 32 + nc]      = (bf16_t)p0s[r];
      pbuf[wave][(r + 8 * hl) * 32 + 16 + nc] = (bf16_t)p1s[r];
    }
    asm volatile("s_wait_dscnt 0" ::: "memory");
    v16bf ap = load_frag_bf16(&pbuf[wave][0], 32, 0, 0, lane);
    asm volatile("" ::: "memory");
    // V as B-operand needs V^T access: element (n=d, k) -> V[k, d]
#pragma unroll
    for (int nn = 0; nn < 4; ++nn) {
      int d = nn * 16 + nc;
      v16bf bv;
#pragma unroll
      for (int e = 0; e < 8; ++e) {
        int k = kc + hl * 8 + e;
        bv[e] = Vx[(size_t)k * (NKV * HDIM) + d];
      }
#pragma unroll
      for (int e = 0; e < 8; ++e) {
        int k = kc + 16 + hl * 8 + e;
        bv[8 + e] = Vx[(size_t)k * (NKV * HDIM) + d];
      }
      o[nn] = wmma_bf16(ap, bv, o[nn]);
    }
  }
  bf16_t* O = ob + (size_t)b * SEQ * (NH * HDIM) + h * HDIM;
#pragma unroll
  for (int nn = 0; nn < 4; ++nn)
#pragma unroll
    for (int r = 0; r < 8; ++r) {
      int m = r + 8 * hl;
      O[(size_t)(qabs + m) * (NH * HDIM) + nn * 16 + nc] =
          (bf16_t)(o[nn][r] / lrow[r]);
    }
}

// ---------------------------------------------------------------------------
// MoE gate: softmax over 8 experts + top-2 + renormalize -> cw[T,8]
// one wave per token
// ---------------------------------------------------------------------------
__global__ void gate_kernel(const bf16_t* __restrict__ hn, const float* __restrict__ gw,
                            float* __restrict__ cw) {
  int lane = threadIdx.x & 31, wave = threadIdx.x >> 5;
  int t = blockIdx.x * 8 + wave;
  float acc[NE];
#pragma unroll
  for (int e = 0; e < NE; ++e) acc[e] = 0.f;
  for (int d = lane; d < DDIM; d += 32) {
    float hv = (float)hn[(size_t)t * DDIM + d];
#pragma unroll
    for (int e = 0; e < NE; ++e) acc[e] += hv * gw[(size_t)e * DDIM + d];
  }
#pragma unroll
  for (int e = 0; e < NE; ++e)
#pragma unroll
    for (int off = 16; off > 0; off >>= 1) acc[e] += __shfl_xor(acc[e], off, 32);
  if (lane == 0) {
    float mx = acc[0];
#pragma unroll
    for (int e = 1; e < NE; ++e) mx = fmaxf(mx, acc[e]);
    float pe[NE], ssum = 0.f;
#pragma unroll
    for (int e = 0; e < NE; ++e) { pe[e] = __expf(acc[e] - mx); ssum += pe[e]; }
#pragma unroll
    for (int e = 0; e < NE; ++e) pe[e] /= ssum;
    int i1 = 0;
#pragma unroll
    for (int e = 1; e < NE; ++e) if (pe[e] > pe[i1]) i1 = e;
    int i2 = (i1 == 0) ? 1 : 0;
#pragma unroll
    for (int e = 0; e < NE; ++e) if (e != i1 && pe[e] > pe[i2]) i2 = e;
    float sw = pe[i1] + pe[i2];
#pragma unroll
    for (int e = 0; e < NE; ++e) cw[(size_t)t * NE + e] = 0.f;
    cw[(size_t)t * NE + i1] = pe[i1] / sw;
    cw[(size_t)t * NE + i2] = pe[i2] / sw;
  }
}

// ---------------------------------------------------------------------------
// MoE expert GEMM 1: inter = silu(A*w1^T) * (A*w3^T), bf16 out, per-expert.
// Block-level skip (all 128 tokens unrouted) + per-wave compute skip; both
// weight tiles LDS-staged, double-buffered; B-frags batched.
// ---------------------------------------------------------------------------
__global__ void moe_h1h3_kernel(const bf16_t* __restrict__ A,
                                const float* __restrict__ W1,
                                const float* __restrict__ W3,
                                const float* __restrict__ cw, int e,
                                bf16_t* __restrict__ inter) {
  __shared__ bf16_t w1t[2][64 * LDW];
  __shared__ bf16_t w3t[2][64 * LDW];
  __shared__ int blk_active;
  int tid = threadIdx.x;
  int lane = tid & 31, wave = tid >> 5;
  int mblk = blockIdx.y * 128;
  int m0 = mblk + wave * 16;
  int n0 = blockIdx.x * 64;

  if (tid == 0) blk_active = 0;
  __syncthreads();
  float fb = (tid < 128) ? cw[(size_t)(mblk + tid) * NE + e] : 0.f;
  if (fb != 0.f) atomicOr(&blk_active, 1);
  __syncthreads();
  if (!blk_active) return;  // uniform: expert unused by all 128 tokens

  float f = (lane < 16) ? cw[(size_t)(m0 + lane) * NE + e] : 0.f;
#pragma unroll
  for (int off = 1; off < 32; off <<= 1) f = fmaxf(f, __shfl_xor(f, off, 32));
  bool wactive = (f != 0.f);

  v8f a1[4] = {v8f_zero(), v8f_zero(), v8f_zero(), v8f_zero()};
  v8f a3[4] = {v8f_zero(), v8f_zero(), v8f_zero(), v8f_zero()};
  float r1[8], r3[8];
  stage_fetch(W1, DDIM, n0, 0, tid, r1);
  stage_fetch(W3, DDIM, n0, 0, tid, r3);
  int nk = DDIM >> 5;
  for (int kt = 0; kt < nk; ++kt) {
    int buf = kt & 1;
    stage_store(&w1t[buf][0], tid, r1);
    stage_store(&w3t[buf][0], tid, r3);
    __syncthreads();
    if (kt + 1 < nk) {
      stage_fetch(W1, DDIM, n0, (kt + 1) * 32, tid, r1);
      stage_fetch(W3, DDIM, n0, (kt + 1) * 32, tid, r3);
    }
    if (wactive) {
      v16bf a = load_frag_bf16(A, DDIM, m0, kt * 32, lane);
      v16bf b10 = load_frag_bf16(&w1t[buf][0], LDW, 0, 0, lane);
      v16bf b11 = load_frag_bf16(&w1t[buf][0], LDW, 16, 0, lane);
      v16bf b12 = load_frag_bf16(&w1t[buf][0], LDW, 32, 0, lane);
      v16bf b13 = load_frag_bf16(&w1t[buf][0], LDW, 48, 0, lane);
      v16bf b30 = load_frag_bf16(&w3t[buf][0], LDW, 0, 0, lane);
      v16bf b31 = load_frag_bf16(&w3t[buf][0], LDW, 16, 0, lane);
      v16bf b32 = load_frag_bf16(&w3t[buf][0], LDW, 32, 0, lane);
      v16bf b33 = load_frag_bf16(&w3t[buf][0], LDW, 48, 0, lane);
      a1[0] = wmma_bf16(a, b10, a1[0]);
      a1[1] = wmma_bf16(a, b11, a1[1]);
      a1[2] = wmma_bf16(a, b12, a1[2]);
      a1[3] = wmma_bf16(a, b13, a1[3]);
      a3[0] = wmma_bf16(a, b30, a3[0]);
      a3[1] = wmma_bf16(a, b31, a3[1]);
      a3[2] = wmma_bf16(a, b32, a3[2]);
      a3[3] = wmma_bf16(a, b33, a3[3]);
    }
  }
  if (!wactive) return;
  int nc = lane & 15, hl = lane >> 4;
#pragma unroll
  for (int nn = 0; nn < 4; ++nn)
#pragma unroll
    for (int r = 0; r < 8; ++r) {
      int m = m0 + r + 8 * hl;
      float x1 = a1[nn][r], x3 = a3[nn][r];
      float sil = x1 / (1.f + __expf(-x1));
      inter[(size_t)m * FDIM + n0 + nn * 16 + nc] = (bf16_t)(sil * x3);
    }
}

// ---------------------------------------------------------------------------
// MoE expert GEMM 2: moe += (inter * w2^T) * cw[:,e]; same skip structure.
// ---------------------------------------------------------------------------
__global__ void moe_out_kernel(const bf16_t* __restrict__ inter,
                               const float* __restrict__ W2,
                               const float* __restrict__ cw, int e,
                               float* __restrict__ moe) {
  __shared__ bf16_t wt[2][64 * LDW];
  __shared__ int blk_active;
  int tid = threadIdx.x;
  int lane = tid & 31, wave = tid >> 5;
  int mblk = blockIdx.y * 128;
  int m0 = mblk + wave * 16;
  int n0 = blockIdx.x * 64;

  if (tid == 0) blk_active = 0;
  __syncthreads();
  float fb = (tid < 128) ? cw[(size_t)(mblk + tid) * NE + e] : 0.f;
  if (fb != 0.f) atomicOr(&blk_active, 1);
  __syncthreads();
  if (!blk_active) return;

  float f = (lane < 16) ? cw[(size_t)(m0 + lane) * NE + e] : 0.f;
#pragma unroll
  for (int off = 1; off < 32; off <<= 1) f = fmaxf(f, __shfl_xor(f, off, 32));
  bool wactive = (f != 0.f);

  v8f acc[4] = {v8f_zero(), v8f_zero(), v8f_zero(), v8f_zero()};
  float rg[8];
  stage_fetch(W2, FDIM, n0, 0, tid, rg);
  int nk = FDIM >> 5;
  for (int kt = 0; kt < nk; ++kt) {
    int buf = kt & 1;
    stage_store(&wt[buf][0], tid, rg);
    __syncthreads();
    if (kt + 1 < nk) stage_fetch(W2, FDIM, n0, (kt + 1) * 32, tid, rg);
    if (wactive) {
      v16bf a = load_frag_bf16(inter, FDIM, m0, kt * 32, lane);
      v16bf b0 = load_frag_bf16(&wt[buf][0], LDW, 0, 0, lane);
      v16bf b1 = load_frag_bf16(&wt[buf][0], LDW, 16, 0, lane);
      v16bf b2 = load_frag_bf16(&wt[buf][0], LDW, 32, 0, lane);
      v16bf b3 = load_frag_bf16(&wt[buf][0], LDW, 48, 0, lane);
      acc[0] = wmma_bf16(a, b0, acc[0]);
      acc[1] = wmma_bf16(a, b1, acc[1]);
      acc[2] = wmma_bf16(a, b2, acc[2]);
      acc[3] = wmma_bf16(a, b3, acc[3]);
    }
  }
  if (!wactive) return;
  int nc = lane & 15, hl = lane >> 4;
#pragma unroll
  for (int nn = 0; nn < 4; ++nn)
#pragma unroll
    for (int r = 0; r < 8; ++r) {
      int m = m0 + r + 8 * hl;
      float scale = cw[(size_t)m * NE + e];
      size_t idx = (size_t)m * DDIM + n0 + nn * 16 + nc;
      moe[idx] += acc[nn][r] * scale;  // experts are serialized launches
    }
}

// ---------------------------------------------------------------------------
extern "C" void kernel_launch(void* const* d_in, const int* in_sizes, int n_in,
                              void* d_out, int out_size, void* d_ws, size_t ws_size,
                              hipStream_t stream) {
  const float* x    = (const float*)d_in[0];
  const float* ln1w = (const float*)d_in[1];
  const float* ln2w = (const float*)d_in[2];
  const float* wq   = (const float*)d_in[3];
  const float* wk   = (const float*)d_in[4];
  const float* wv   = (const float*)d_in[5];
  const float* wo   = (const float*)d_in[6];
  const float* gw   = (const float*)d_in[7];
  const float* w1   = (const float*)d_in[8];
  const float* w2   = (const float*)d_in[9];
  const float* w3   = (const float*)d_in[10];
  const float* cosb = (const float*)d_in[11];
  const float* sinb = (const float*)d_in[12];
  float* out = (float*)d_out;

  char* wsp = (char*)d_ws;
  auto alloc = [&](size_t bytes) {
    char* p = wsp;
    wsp += (bytes + 255) & ~(size_t)255;
    return (void*)p;
  };
  bf16_t* hn1   = (bf16_t*)alloc((size_t)TTOK * DDIM * 2);
  float*  qf    = (float*) alloc((size_t)TTOK * NH  * HDIM * 4);
  float*  kf    = (float*) alloc((size_t)TTOK * NKV * HDIM * 4);
  float*  vf    = (float*) alloc((size_t)TTOK * NKV * HDIM * 4);
  bf16_t* qb    = (bf16_t*)alloc((size_t)TTOK * NH  * HDIM * 2);
  bf16_t* kbuf  = (bf16_t*)alloc((size_t)TTOK * NKV * HDIM * 2);
  bf16_t* vbuf  = (bf16_t*)alloc((size_t)TTOK * NKV * HDIM * 2);
  bf16_t* attnb = (bf16_t*)alloc((size_t)TTOK * NH  * HDIM * 2);
  float*  hres  = (float*) alloc((size_t)TTOK * DDIM * 4);
  bf16_t* hn2   = (bf16_t*)alloc((size_t)TTOK * DDIM * 2);
  float*  cw    = (float*) alloc((size_t)TTOK * NE * 4);
  bf16_t* inter = (bf16_t*)alloc((size_t)TTOK * FDIM * 2);
  float*  moe   = (float*) alloc((size_t)TTOK * DDIM * 4);

  dim3 blk(256);

  // 1. RMSNorm 1
  rmsnorm_kernel<<<TTOK, blk, 0, stream>>>(x, ln1w, hn1);
  // 2. QKV projections
  gemm_kernel<<<dim3(NH * HDIM / 64, TTOK / 128), blk, 0, stream>>>(hn1, wq, qf, DDIM, NH * HDIM);
  gemm_kernel<<<dim3(NKV * HDIM / 64, TTOK / 128), blk, 0, stream>>>(hn1, wk, kf, DDIM, NKV * HDIM);
  gemm_kernel<<<dim3(NKV * HDIM / 64, TTOK / 128), blk, 0, stream>>>(hn1, wv, vf, DDIM, NKV * HDIM);
  // 3. RoPE + bf16 convert
  rope_kernel<<<(TTOK * NH * 32 + 255) / 256, blk, 0, stream>>>(qf, cosb, sinb, qb, NH);
  rope_kernel<<<(TTOK * NKV * 32 + 255) / 256, blk, 0, stream>>>(kf, cosb, sinb, kbuf, NKV);
  cvt_kernel<<<(TTOK * NKV * HDIM + 255) / 256, blk, 0, stream>>>(vf, vbuf, TTOK * NKV * HDIM);
  // 4. Flash attention
  attn_kernel<<<dim3(SEQ / 128, NH, 2), blk, 0, stream>>>(qb, kbuf, vbuf, attnb);
  // 5. O-projection + residual
  gemm_residual_kernel<<<dim3(DDIM / 64, TTOK / 128), blk, 0, stream>>>(attnb, wo, x, hres, NH * HDIM, DDIM);
  // 6. RMSNorm 2
  rmsnorm_kernel<<<TTOK, blk, 0, stream>>>(hres, ln2w, hn2);
  // 7. Gate (softmax + top-2)
  gate_kernel<<<TTOK / 8, blk, 0, stream>>>(hn2, gw, cw);
  // 8. MoE experts (serialized, tile-skipped by routing weights)
  zero_kernel<<<(TTOK * DDIM + 255) / 256, blk, 0, stream>>>(moe, TTOK * DDIM);
  for (int e = 0; e < NE; ++e) {
    const float* W1e = w1 + (size_t)e * FDIM * DDIM;
    const float* W3e = w3 + (size_t)e * FDIM * DDIM;
    const float* W2e = w2 + (size_t)e * DDIM * FDIM;
    moe_h1h3_kernel<<<dim3(FDIM / 64, TTOK / 128), blk, 0, stream>>>(hn2, W1e, W3e, cw, e, inter);
    moe_out_kernel<<<dim3(DDIM / 64, TTOK / 128), blk, 0, stream>>>(inter, W2e, cw, e, moe);
  }
  // 9. Final residual
  addout_kernel<<<(TTOK * DDIM + 255) / 256, blk, 0, stream>>>(hres, moe, out, TTOK * DDIM);
}
